// VectorQuantizerEMA_47382079209956
// MI455X (gfx1250) — matrix operand order repaired and checked by reading, not compiled, not run on hardware
//
#include <hip/hip_runtime.h>
#include <hip/hip_bf16.h>
#include <math.h>

// ---------------------------------------------------------------------------
// Problem constants: z[64,256,32,32], embedding[1024,256]
// ---------------------------------------------------------------------------
#define BB 64
#define DD 256
#define HW 1024          // H*W = 32*32
#define NN (BB * HW)     // 65536 flattened vectors
#define KK 1024

// d_out layout (floats), reference return order
#define ZQ_OFF   0
#define ZQ_N     (BB * DD * HW)           // 16777216
#define LOSS_OFF (ZQ_OFF + ZQ_N)          // 16777216
#define IDX_OFF  (LOSS_OFF + 1)           // 16777217
#define PERP_OFF (IDX_OFF + NN)           // 16842753
#define USED_OFF (PERP_OFF + 1)           // 16842754
#define EMB_OFF  (USED_OFF + KK)          // 16843778
#define CSN_OFF  (EMB_OFF + KK * DD)      // 17105922
#define EAN_OFF  (CSN_OFF + KK)           // 17106946

// workspace layout (bytes)
#define WS_EMB16  0                        // K*D f16            = 524288
#define WS_ENORM  524288                   // K f32              = 4096
#define WS_IDX    528384                   // N int32            = 262144
#define WS_COUNTS 790528                   // K f32              = 4096
#define WS_ESUM   794624                   // K*D f32            = 1048576
#define WS_LOSS   1843200                  // 1 f32 (+pad)       = 256
#define WS_ZERO_OFF  WS_COUNTS
#define WS_ZERO_LEN  (4096 + 1048576 + 256)

typedef _Float16 v8h  __attribute__((ext_vector_type(8)));
typedef _Float16 v16h __attribute__((ext_vector_type(16)));
typedef float    v8f  __attribute__((ext_vector_type(8)));

// ---------------------------------------------------------------------------
// Kernel 1: embedding f32 -> f16 (row major) + per-code squared L2 norm
// ---------------------------------------------------------------------------
__global__ __launch_bounds__(256) void vq_prep_kernel(
    const float* __restrict__ emb, _Float16* __restrict__ emb16,
    float* __restrict__ enorm)
{
  __shared__ float red[256];
  const int k = blockIdx.x;        // code
  const int d = threadIdx.x;       // feature
  float v = emb[k * DD + d];
  emb16[k * DD + d] = (_Float16)v;
  red[d] = v * v;
  __syncthreads();
  for (int s = 128; s > 0; s >>= 1) {
    if (d < s) red[d] += red[d + s];
    __syncthreads();
  }
  if (d == 0) enorm[k] = red[0];
}

// ---------------------------------------------------------------------------
// Kernel 2: WMMA argmin over the codebook.
//   Tile = 64 consecutive flattened rows (one b, 64 consecutive hw).
//   8 waves: wave = (rowgrp 0..3) x (code-half 0..1).
//   dist(row,code) = ||e||^2 - 2 * <z_row, e_code>   (||z||^2 is row-constant)
// ---------------------------------------------------------------------------
#define TROWS 64
#define DPAD  264   // f16 elements per LDS row (16B padding -> bank spread)

__global__ __launch_bounds__(256) void vq_argmin_kernel(
    const float* __restrict__ z, const _Float16* __restrict__ emb16,
    const float* __restrict__ enorm, int* __restrict__ idx_ws,
    float* __restrict__ counts, float* __restrict__ idx_out)
{
  __shared__ _Float16 zt[TROWS * DPAD];   // 33792 B
  __shared__ float    mv[TROWS * 2];
  __shared__ int      mi[TROWS * 2];

  const int n0  = blockIdx.x * TROWS;     // first flattened row of tile
  const int b   = n0 >> 10;               // / HW
  const int hw0 = n0 & (HW - 1);
  const float* zb = z + (size_t)b * (DD * HW);

  // stage z tile (f32 -> f16), reads coalesced across hw
  {
    const int hw = threadIdx.x & 63;
    const int d0 = threadIdx.x >> 6;      // 0..3
    for (int d = d0; d < DD; d += 4)
      zt[hw * DPAD + d] = (_Float16)zb[d * HW + hw0 + hw];
  }
  __syncthreads();

  const int wave   = threadIdx.x >> 5;
  const int lane   = threadIdx.x & 31;
  const int rowgrp = wave & 3;            // which 16-row group
  const int half   = wave >> 2;           // which 512-code half
  const int l16    = lane & 15;
  const int h      = lane >> 4;           // lane half (ISA K-swizzle select)

  // A fragments for this wave's 16 rows, all of D (8 k-steps of 32).
  // A layout (matches B): lane half h holds contiguous K=16h..16h+15.
  v16h a[8];
  const _Float16* arow = zt + (rowgrp * 16 + l16) * DPAD;
  #pragma unroll
  for (int ks = 0; ks < 8; ++ks) {
    v8h lo = *(const v8h*)(arow + ks * 32 + 16 * h);
    v8h hi = *(const v8h*)(arow + ks * 32 + 16 * h + 8);
    a[ks] = __builtin_shufflevector(lo, hi,
              0,1,2,3,4,5,6,7,8,9,10,11,12,13,14,15);
  }

  float minv[8];
  int   mini[8];
  #pragma unroll
  for (int r = 0; r < 8; ++r) { minv[r] = 3.0e38f; mini[r] = 0; }

  const int cbase = half * 512 + l16;
  for (int ct = 0; ct < 32; ++ct) {
    const int code = cbase + ct * 16;
    // B layout (same as A): lane half h holds contiguous K = ksbase+16h..+15
    const _Float16* brow = emb16 + code * DD + 16 * h;
    if (ct + 1 < 32)
      __builtin_prefetch(emb16 + (code + 16) * DD, 0, 3);
    v8f acc = {};
    #pragma unroll
    for (int ks = 0; ks < 8; ++ks) {
      v16h bv = *(const v16h*)(brow + ks * 32);
      acc = __builtin_amdgcn_wmma_f32_16x16x32_f16(
              false, a[ks], false, bv, (short)0, acc, false, false);
    }
    const float en = enorm[code];
    // C layout: VGPR r on lane-half h -> row M = r + 8h, col N = lane&15
    #pragma unroll
    for (int r = 0; r < 8; ++r) {
      float dist = en - 2.0f * acc[r];
      if (dist < minv[r]) { minv[r] = dist; mini[r] = code; }
    }
  }

  // reduce the 16 code columns held across lanes of each lane-half
  #pragma unroll
  for (int m = 8; m >= 1; m >>= 1) {
    #pragma unroll
    for (int r = 0; r < 8; ++r) {
      float ov = __shfl_xor(minv[r], m, 32);
      int   oi = __shfl_xor(mini[r], m, 32);
      if (ov < minv[r] || (ov == minv[r] && oi < mini[r])) {
        minv[r] = ov; mini[r] = oi;
      }
    }
  }
  if (l16 == 0) {
    #pragma unroll
    for (int r = 0; r < 8; ++r) {
      const int row_local = rowgrp * 16 + 8 * h + r;
      mv[row_local * 2 + half] = minv[r];
      mi[row_local * 2 + half] = mini[r];
    }
  }
  __syncthreads();

  // merge the two code halves, emit index + count
  if (threadIdx.x < TROWS) {
    const int rl = threadIdx.x;
    const float v0 = mv[rl * 2 + 0], v1 = mv[rl * 2 + 1];
    const int   i0 = mi[rl * 2 + 0], i1 = mi[rl * 2 + 1];
    const int best = (v1 < v0) ? i1 : i0;
    const int n = n0 + rl;
    idx_ws[n] = best;
    idx_out[n] = (float)best;
    atomicAdd(&counts[best], 1.0f);
  }
}

// ---------------------------------------------------------------------------
// Kernel 3: fused scatter (embed_sum), gather (quantized), z_q write, loss.
//   Block = 32 hw rows of one b. z staged in LDS so both global directions
//   stay coalesced; embedding gathers are full contiguous rows.
// ---------------------------------------------------------------------------
#define QROWS 32
#define QPAD  265   // stride 265 f32 -> bank stride 9, conflict-free transpose

__global__ __launch_bounds__(256) void vq_quant_kernel(
    const float* __restrict__ z, const float* __restrict__ emb,
    const int* __restrict__ idx_ws, float* __restrict__ embed_sum,
    float* __restrict__ zq_out, float* __restrict__ loss_acc)
{
  __shared__ float zt[QROWS * QPAD];      // 33920 B
  __shared__ int   idxsh[QROWS];
  __shared__ float red[256];

  const int b   = blockIdx.x >> 5;        // 32 chunks per b
  const int hw0 = (blockIdx.x & 31) * QROWS;
  const float* zb = z + (size_t)b * (DD * HW);
  float*       qb = zq_out + (size_t)b * (DD * HW);
  const int n0 = b * HW + hw0;

  // stage z tile [row=hw][d], reads coalesced across hw
  {
    const int hw = threadIdx.x & 31;
    const int d0 = threadIdx.x >> 5;      // 0..7
    for (int d = d0; d < DD; d += 8)
      zt[hw * QPAD + d] = zb[d * HW + hw0 + hw];
  }
  if (threadIdx.x < QROWS) idxsh[threadIdx.x] = idx_ws[n0 + threadIdx.x];
  __syncthreads();

  const int d = threadIdx.x;              // 0..255 = feature
  float lacc = 0.0f;
  for (int r = 0; r < QROWS; ++r) {
    const int code = idxsh[r];
    const float zv = zt[r * QPAD + d];
    atomicAdd(&embed_sum[code * DD + d], zv);      // segment_sum(flat_z)
    const float q = emb[code * DD + d];            // coalesced row gather
    const float diff = q - zv;
    lacc += diff * diff;
    zt[r * QPAD + d] = q;                          // overwrite with quantized
  }
  __syncthreads();

  // write z_q back in [B,D,H,W] order, coalesced across hw
  {
    const int hw = threadIdx.x & 31;
    const int d0 = threadIdx.x >> 5;
    for (int dd = d0; dd < DD; dd += 8)
      qb[dd * HW + hw0 + hw] = zt[hw * QPAD + dd];
  }

  red[threadIdx.x] = lacc;
  __syncthreads();
  for (int s = 128; s > 0; s >>= 1) {
    if (threadIdx.x < s) red[threadIdx.x] += red[threadIdx.x + s];
    __syncthreads();
  }
  if (threadIdx.x == 0) atomicAdd(loss_acc, red[0]);
}

// ---------------------------------------------------------------------------
// Kernel 4: EMA updates, smoothing, perplexity, used codes, loss finalize.
// ---------------------------------------------------------------------------
__global__ __launch_bounds__(256) void vq_final_kernel(
    const float* __restrict__ cs, const float* __restrict__ ea,
    const float* __restrict__ counts, const float* __restrict__ esum,
    const float* __restrict__ loss_acc, float* __restrict__ out)
{
  __shared__ float red[256];
  __shared__ float s_n, s_cnt;
  const int t = threadIdx.x;

  float accn = 0.0f, accc = 0.0f;
  for (int k = t; k < KK; k += 256) {
    const float cnt = counts[k];
    const float csn = 0.99f * cs[k] + 0.01f * cnt;
    out[CSN_OFF + k]  = csn;
    out[USED_OFF + k] = (cnt > 0.0f) ? 1.0f : 0.0f;
    accn += csn;
    accc += cnt;
  }
  red[t] = accn; __syncthreads();
  for (int s = 128; s > 0; s >>= 1) { if (t < s) red[t] += red[t + s]; __syncthreads(); }
  if (t == 0) s_n = red[0];
  __syncthreads();
  red[t] = accc; __syncthreads();
  for (int s = 128; s > 0; s >>= 1) { if (t < s) red[t] += red[t + s]; __syncthreads(); }
  if (t == 0) s_cnt = red[0];
  __syncthreads();

  float ent = 0.0f;
  for (int k = t; k < KK; k += 256) {
    const float p = counts[k] / s_cnt;
    ent += p * logf(p + 1e-10f);
  }
  red[t] = ent; __syncthreads();
  for (int s = 128; s > 0; s >>= 1) { if (t < s) red[t] += red[t + s]; __syncthreads(); }
  if (t == 0) {
    out[PERP_OFF] = expf(-red[0]);
    out[LOSS_OFF] = loss_acc[0] * (1.0f / (float)ZQ_N);
  }
  __syncthreads();

  const float nsum  = s_n;
  const float denom = nsum + (float)KK * 1e-5f;
  for (int m = t; m < KK * DD; m += 256) {
    const int k = m >> 8;                  // / DD
    const float ean = 0.99f * ea[m] + 0.01f * esum[m];
    out[EAN_OFF + m] = ean;
    const float csn = out[CSN_OFF + k];
    const float sm  = (csn + 1e-5f) / denom * nsum;
    out[EMB_OFF + m] = ean / sm;
  }
}

// ---------------------------------------------------------------------------
// Host launcher
// ---------------------------------------------------------------------------
extern "C" void kernel_launch(void* const* d_in, const int* in_sizes, int n_in,
                              void* d_out, int out_size, void* d_ws, size_t ws_size,
                              hipStream_t stream)
{
  (void)in_sizes; (void)n_in; (void)out_size; (void)ws_size;
  const float* z   = (const float*)d_in[0];   // [64,256,32,32]
  const float* emb = (const float*)d_in[1];   // [1024,256]
  const float* cs  = (const float*)d_in[2];   // [1024]
  const float* ea  = (const float*)d_in[3];   // [1024,256]
  float* out = (float*)d_out;
  char*  ws  = (char*)d_ws;

  _Float16* emb16  = (_Float16*)(ws + WS_EMB16);
  float*    enorm  = (float*)(ws + WS_ENORM);
  int*      idxw   = (int*)(ws + WS_IDX);
  float*    counts = (float*)(ws + WS_COUNTS);
  float*    esum   = (float*)(ws + WS_ESUM);
  float*    lossa  = (float*)(ws + WS_LOSS);

  // zero accumulators (counts, embed_sum, loss) -- graph-capture safe
  hipMemsetAsync(ws + WS_ZERO_OFF, 0, WS_ZERO_LEN, stream);

  vq_prep_kernel<<<KK, 256, 0, stream>>>(emb, emb16, enorm);
  vq_argmin_kernel<<<NN / TROWS, 256, 0, stream>>>(z, emb16, enorm, idxw,
                                                   counts, out + IDX_OFF);
  vq_quant_kernel<<<(BB * HW) / QROWS, 256, 0, stream>>>(z, emb, idxw, esum,
                                                         out + ZQ_OFF, lossa);
  vq_final_kernel<<<1, 256, 0, stream>>>(cs, ea, counts, esum, lossa, out);
}
